// TransformerLSTMWithSelfAttention_71700184040095
// MI455X (gfx1250) — compile-verified, matrix operand order checked by reading, not run on hardware
//
#include <hip/hip_runtime.h>
#include <cstdint>
#include <cstddef>

// ---------------------------------------------------------------------------
// Types for CDNA5 WMMA (wave32): v_wmma_f32_16x16x32_bf16
// ---------------------------------------------------------------------------
typedef __bf16 bf16_t;
typedef __attribute__((ext_vector_type(16))) __bf16       bf16x16;
typedef __attribute__((ext_vector_type(8)))  float        f32x8;
typedef __attribute__((ext_vector_type(4)))  unsigned int u32x4;

#define B_   256
#define T_   128
#define IN_  512
#define H_   1024
#define NH_  4
#define HD_  256
#define OUT_ 128
#define L_   2

#define RS   40   // LDS tile row stride (bf16 elems): 80B -> 16B aligned, bank-conflict free
#define TM   256  // GEMM block tile M
#define TN   128  // GEMM block tile N

// ---------------------------------------------------------------------------
// WMMA helpers
// ---------------------------------------------------------------------------
__device__ __forceinline__ bf16x16 load_frag(const bf16_t* tile, int stride) {
  // 16-bit A/B fragment per ISA 7.12.2: lane<16 holds K={0..7,16..23} of row (lane&15),
  // lane>=16 holds K={8..15,24..31}. Two 16B ds_load_b128 per lane.
  const int lane = threadIdx.x & 31;
  const int s = (lane >> 4) << 3;                  // 0 or 8
  const bf16_t* p = tile + (lane & 15) * stride;
  union { bf16x16 v; u32x4 q[2]; } f;
  f.q[0] = *(const u32x4*)(p + s);                 // K = s .. s+7
  f.q[1] = *(const u32x4*)(p + s + 16);            // K = 16+s .. 16+s+7
  return f.v;
}

__device__ __forceinline__ f32x8 wmma_bf16(bf16x16 a, bf16x16 b, f32x8 c) {
  // (neg_a, A, neg_b, B, c_mod, C, reuse_a, reuse_b)
  return __builtin_amdgcn_wmma_f32_16x16x32_bf16(false, a, false, b, (short)0, c, false, false);
}

__device__ __forceinline__ f32x8 zero8() {
  return (f32x8){0.f, 0.f, 0.f, 0.f, 0.f, 0.f, 0.f, 0.f};
}

__device__ __forceinline__ float sigf(float x) { return 1.0f / (1.0f + expf(-x)); }

// ---------------------------------------------------------------------------
// fp32 -> bf16 conversion (grid-stride)
// ---------------------------------------------------------------------------
__global__ void cvt_k(const float* __restrict__ s, bf16_t* __restrict__ d, size_t n) {
  size_t i = (size_t)blockIdx.x * blockDim.x + threadIdx.x;
  const size_t stride = (size_t)gridDim.x * blockDim.x;
  for (; i < n; i += stride) d[i] = (bf16_t)s[i];
}

// ---------------------------------------------------------------------------
// Generic WMMA GEMM: C[M,N] = act( A[M,K](bf16) @ W[N,K]^T(bf16) + b0 + b1 )
//   Cf (fp32) and/or Cb (bf16) outputs; permBT remaps bf16 row b*T+t -> t*B+b.
//   Requires M%256==0, N%128==0, K%32==0 (true for every call here).
// Block: 256 threads = 8 waves; 256x128 output tile; wave = 64x64 sub-tile
// (16 accumulators, 4 A-frags + 4 B-frags per K chunk -> 16 WMMA / 8 frag loads).
// Register double-buffer: next K-tile global loads issued before the WMMA block.
// ---------------------------------------------------------------------------
__global__ __launch_bounds__(256) void gemm_bf16_k(
    const bf16_t* __restrict__ A, const bf16_t* __restrict__ W,
    const float* __restrict__ b0, const float* __restrict__ b1,
    float* __restrict__ Cf, bf16_t* __restrict__ Cb,
    int M, int N, int K, int relu, int permBT)
{
  __shared__ bf16_t As[TM * RS];
  __shared__ bf16_t Ws[TN * RS];
  const int tid  = threadIdx.x;
  const int lane = tid & 31;
  const int wave = tid >> 5;
  const int wm = wave >> 1;                       // 0..3  (64-row strip)
  const int wn = wave & 1;                        // 0..1  (64-col strip)
  const size_t m0 = (size_t)blockIdx.y * TM;
  const size_t n0 = (size_t)blockIdx.x * TN;

  f32x8 acc[4][4];
#pragma unroll
  for (int i = 0; i < 4; ++i)
#pragma unroll
    for (int j = 0; j < 4; ++j) acc[i][j] = zero8();

  // A staging: thread -> one of 256 rows, full 32-elem (64B) K slice.
  const bf16_t* ga = A + (m0 + tid) * (size_t)K;
  // W staging: thread pair -> one of 128 rows, 16-elem half slice.
  const int wrow = tid >> 1, wcol = (tid & 1) << 4;
  const bf16_t* gw = W + (n0 + wrow) * (size_t)K + wcol;

  u32x4 ra0 = *(const u32x4*)(ga + 0);
  u32x4 ra1 = *(const u32x4*)(ga + 8);
  u32x4 ra2 = *(const u32x4*)(ga + 16);
  u32x4 ra3 = *(const u32x4*)(ga + 24);
  u32x4 rw0 = *(const u32x4*)(gw + 0);
  u32x4 rw1 = *(const u32x4*)(gw + 8);

  for (int k0 = 0; k0 < K; k0 += 32) {
    __syncthreads();                               // LDS reads of prev chunk done
    *(u32x4*)&As[tid * RS + 0]         = ra0;
    *(u32x4*)&As[tid * RS + 8]         = ra1;
    *(u32x4*)&As[tid * RS + 16]        = ra2;
    *(u32x4*)&As[tid * RS + 24]        = ra3;
    *(u32x4*)&Ws[wrow * RS + wcol]     = rw0;
    *(u32x4*)&Ws[wrow * RS + wcol + 8] = rw1;
    __syncthreads();

    if (k0 + 32 < K) {                             // issue next tile early; wait lands
      ra0 = *(const u32x4*)(ga + k0 + 32);         // after the WMMA block below
      ra1 = *(const u32x4*)(ga + k0 + 40);
      ra2 = *(const u32x4*)(ga + k0 + 48);
      ra3 = *(const u32x4*)(ga + k0 + 56);
      rw0 = *(const u32x4*)(gw + k0 + 32);
      rw1 = *(const u32x4*)(gw + k0 + 40);
      if (k0 + 64 < K) {                           // near-cache prefetch, 2 tiles ahead
        __builtin_prefetch(ga + k0 + 64, 0, 3);
        __builtin_prefetch(gw + k0 + 64, 0, 3);
      }
    }

    bf16x16 af[4], wf[4];
#pragma unroll
    for (int i = 0; i < 4; ++i) af[i] = load_frag(&As[(wm * 64 + i * 16) * RS], RS);
#pragma unroll
    for (int j = 0; j < 4; ++j) wf[j] = load_frag(&Ws[(wn * 64 + j * 16) * RS], RS);
#pragma unroll
    for (int i = 0; i < 4; ++i)
#pragma unroll
      for (int j = 0; j < 4; ++j)
        acc[i][j] = wmma_bf16(af[i], wf[j], acc[i][j]);
  }

  // Epilogue. C/D layout: lane<16 -> N=lane, M=r; lane>=16 -> N=lane-16, M=8+r.
  const int nl = lane & 15;
  const int mh = (lane >> 4) << 3;
#pragma unroll
  for (int i = 0; i < 4; ++i) {
#pragma unroll
    for (int j = 0; j < 4; ++j) {
      const size_t gn = n0 + wn * 64 + j * 16 + nl;
      float bias = 0.f;
      if (b0) bias += b0[gn];
      if (b1) bias += b1[gn];
#pragma unroll
      for (int r = 0; r < 8; ++r) {
        const size_t gm = m0 + wm * 64 + i * 16 + mh + r;
        float v = acc[i][j][r] + bias;
        if (relu) v = fmaxf(v, 0.f);
        if (Cf) Cf[gm * (size_t)N + gn] = v;
        if (Cb) {
          size_t orow = gm;
          if (permBT) {                            // row b*T+t -> t*B+b
            const size_t bb = gm / T_;
            const size_t tt = gm - bb * T_;
            orow = tt * B_ + bb;
          }
          Cb[orow * (size_t)N + gn] = (bf16_t)v;
        }
      }
    }
  }
}

// ---------------------------------------------------------------------------
// Causal multi-head attention, one workgroup per (b,h).
// qkv: [B*T, 3H] bf16 (q | k | v); ctx: [B*T, H] bf16.
// Dynamic LDS (188KB, legal on CDNA5's 320KB/WG):
//   [0..20480)      q,k tiles (128x40 bf16 each)
//   [20480..88064)  S fp32 [128][132]
//   [88064..122880) P bf16 [128][136]
//   [122880..192512) V^T bf16 [256][136]
// ---------------------------------------------------------------------------
#define ATTN_SMEM 192512

__global__ __launch_bounds__(256) void attn_k(const bf16_t* __restrict__ qkv,
                                              bf16_t* __restrict__ ctx)
{
  extern __shared__ char smem[];
  bf16_t* qs = (bf16_t*)smem;                     // 128*40
  bf16_t* ks = qs + 128 * RS;                     // 128*40
  float*  Sf = (float*)(smem + 20480);            // 128*132
  bf16_t* Pm = (bf16_t*)(smem + 20480 + 128 * 132 * 4);  // 128*136
  bf16_t* Vt = Pm + 128 * 136;                    // 256*136

  const int bh = blockIdx.x;
  const int b  = bh >> 2;
  const int h  = bh & 3;
  const int tid  = threadIdx.x;
  const int lane = tid & 31;
  const int wave = tid >> 5;
  const int wm = wave >> 2, wn = wave & 3;
  const int ldrow = tid >> 1, ldcol = (tid & 1) << 4;

  const size_t base = ((size_t)b * T_) * (3 * H_) + (size_t)h * HD_;

  // ---- Phase 1: S = q @ k^T  (M=N=T=128, K=HD=256), reg double-buffered ----
  f32x8 acc[4][2];
#pragma unroll
  for (int i = 0; i < 4; ++i)
#pragma unroll
    for (int j = 0; j < 2; ++j) acc[i][j] = zero8();

  const bf16_t* gq = qkv + base + (size_t)ldrow * (3 * H_) + ldcol;
  const bf16_t* gk = gq + H_;

  u32x4 rq0 = *(const u32x4*)(gq + 0);
  u32x4 rq1 = *(const u32x4*)(gq + 8);
  u32x4 rk0 = *(const u32x4*)(gk + 0);
  u32x4 rk1 = *(const u32x4*)(gk + 8);

  for (int k0 = 0; k0 < HD_; k0 += 32) {
    __syncthreads();
    *(u32x4*)&qs[ldrow * RS + ldcol]     = rq0;
    *(u32x4*)&qs[ldrow * RS + ldcol + 8] = rq1;
    *(u32x4*)&ks[ldrow * RS + ldcol]     = rk0;
    *(u32x4*)&ks[ldrow * RS + ldcol + 8] = rk1;
    __syncthreads();
    if (k0 + 32 < HD_) {
      rq0 = *(const u32x4*)(gq + k0 + 32);
      rq1 = *(const u32x4*)(gq + k0 + 40);
      rk0 = *(const u32x4*)(gk + k0 + 32);
      rk1 = *(const u32x4*)(gk + k0 + 40);
    }
    bf16x16 af[4], wf[2];
#pragma unroll
    for (int i = 0; i < 4; ++i) af[i] = load_frag(&qs[(wm * 64 + i * 16) * RS], RS);
#pragma unroll
    for (int j = 0; j < 2; ++j) wf[j] = load_frag(&ks[(wn * 32 + j * 16) * RS], RS);
#pragma unroll
    for (int i = 0; i < 4; ++i)
#pragma unroll
      for (int j = 0; j < 2; ++j)
        acc[i][j] = wmma_bf16(af[i], wf[j], acc[i][j]);
  }

  // scale + causal mask -> Sf
  const int nl = lane & 15;
  const int mh = (lane >> 4) << 3;
#pragma unroll
  for (int i = 0; i < 4; ++i)
#pragma unroll
    for (int j = 0; j < 2; ++j)
#pragma unroll
      for (int r = 0; r < 8; ++r) {
        const int m = wm * 64 + i * 16 + mh + r;
        const int n = wn * 32 + j * 16 + nl;
        Sf[m * 132 + n] = (n <= m) ? acc[i][j][r] * 0.0625f : -1e30f;  // 1/sqrt(256)
      }
  __syncthreads();

  // ---- Phase 2: row softmax (f32) -> P (bf16) ----
  if (tid < 128) {
    float mx = -3.4e38f;
    for (int c = 0; c < 128; ++c) mx = fmaxf(mx, Sf[tid * 132 + c]);
    float ssum = 0.f;
    for (int c = 0; c < 128; ++c) ssum += expf(Sf[tid * 132 + c] - mx);
    const float inv = 1.f / ssum;
    for (int c = 0; c < 128; ++c)
      Pm[tid * 136 + c] = (bf16_t)(expf(Sf[tid * 132 + c] - mx) * inv);
  }
  // stage V^T: Vt[d][t] = v[t][d]   (d = tid, coalesced global reads)
  for (int i = 0; i < T_; ++i)
    Vt[tid * 136 + i] = qkv[base + 2 * H_ + (size_t)i * (3 * H_) + tid];
  __syncthreads();

  // ---- Phase 3: ctx = P @ V  (M=128, K=128, N=256); wave -> 16-row strip ----
  for (int nt = 0; nt < 16; ++nt) {
    f32x8 a = zero8();
#pragma unroll
    for (int kc = 0; kc < 4; ++kc) {
      bf16x16 pa = load_frag(Pm + wave * 16 * 136 + kc * 32, 136);
      bf16x16 vb = load_frag(Vt + nt * 16 * 136 + kc * 32, 136);
      a = wmma_bf16(pa, vb, a);
    }
    const int d = nt * 16 + nl;
#pragma unroll
    for (int r = 0; r < 8; ++r) {
      const int t = wave * 16 + mh + r;
      ctx[((size_t)(b * T_ + t)) * H_ + h * HD_ + d] = (bf16_t)a[r];
    }
  }
}

// ---------------------------------------------------------------------------
// LSTM state init: h0=game.reshape(L,B,H)[l], c0=user.reshape(L,B,H)[l]
// ---------------------------------------------------------------------------
__global__ __launch_bounds__(256) void lstm_init_k(const float* __restrict__ game,
                                                   const float* __restrict__ user,
                                                   bf16_t* __restrict__ hbf,
                                                   float* __restrict__ c, int l)
{
  const int idx = blockIdx.x * 256 + threadIdx.x;   // < B*H
  const int b = idx >> 10, j = idx & 1023;
  const size_t src = ((size_t)(l * B_ + b)) * H_ + j;
  hbf[idx] = (bf16_t)game[src];
  c[idx]   = user[src];
}

// ---------------------------------------------------------------------------
// LSTM pointwise: gates = xg[t*B+b] + hg[b]; torch order i,f,g,o
// ---------------------------------------------------------------------------
__global__ __launch_bounds__(256) void lstm_point_k(
    const float* __restrict__ xg, const float* __restrict__ hg,
    float* __restrict__ c, bf16_t* __restrict__ hbf, bf16_t* __restrict__ seqout,
    float* __restrict__ foutH, float* __restrict__ foutC, int t)
{
  const int idx = blockIdx.x * 256 + threadIdx.x;   // < B*H
  const int b = idx >> 10, j = idx & 1023;
  const size_t ro = ((size_t)(t * B_ + b)) * (4 * H_);
  const size_t rh = (size_t)b * (4 * H_);
  const float gi = xg[ro + j]            + hg[rh + j];
  const float gf = xg[ro + H_ + j]       + hg[rh + H_ + j];
  const float gg = xg[ro + 2 * H_ + j]   + hg[rh + 2 * H_ + j];
  const float go = xg[ro + 3 * H_ + j]   + hg[rh + 3 * H_ + j];
  const float cp = c[idx];
  const float cn = sigf(gf) * cp + sigf(gi) * tanhf(gg);
  const float hn = sigf(go) * tanhf(cn);
  c[idx]   = cn;
  hbf[idx] = (bf16_t)hn;
  seqout[((size_t)(t * B_ + b)) * H_ + j] = (bf16_t)hn;
  if (foutH) { foutH[idx] = hn; foutC[idx] = cn; }
}

// ---------------------------------------------------------------------------
// log_softmax over last dim; rows t*B+b -> output [B,T,OUT]
// ---------------------------------------------------------------------------
__global__ __launch_bounds__(128) void logsoftmax_k(const float* __restrict__ logits,
                                                    float* __restrict__ out)
{
  __shared__ float red[128];
  const int row = blockIdx.x;               // t*B + b
  const int cix = threadIdx.x;
  const float v = logits[(size_t)row * OUT_ + cix];
  red[cix] = v;
  __syncthreads();
  for (int s = 64; s > 0; s >>= 1) {
    if (cix < s) red[cix] = fmaxf(red[cix], red[cix + s]);
    __syncthreads();
  }
  const float mx = red[0];
  __syncthreads();
  red[cix] = expf(v - mx);
  __syncthreads();
  for (int s = 64; s > 0; s >>= 1) {
    if (cix < s) red[cix] += red[cix + s];
    __syncthreads();
  }
  const float lse = logf(red[0]) + mx;
  const int t = row / B_, b = row % B_;
  out[((size_t)(b * T_ + t)) * OUT_ + cix] = v - lse;
}

// ---------------------------------------------------------------------------
// Host orchestration
// ---------------------------------------------------------------------------
extern "C" void kernel_launch(void* const* d_in, const int* in_sizes, int n_in,
                              void* d_out, int out_size, void* d_ws, size_t ws_size,
                              hipStream_t stream) {
  (void)in_sizes; (void)n_in; (void)out_size; (void)ws_size;
  const float* x        = (const float*)d_in[0];
  const float* user_vec = (const float*)d_in[1];
  const float* game_vec = (const float*)d_in[2];
  const float* fc_w     = (const float*)d_in[3];
  const float* fc_b     = (const float*)d_in[4];
  const float* ipw      = (const float*)d_in[5];
  const float* ipb      = (const float*)d_in[6];
  const float* opw      = (const float*)d_in[7];
  const float* opb      = (const float*)d_in[8];
  const float* w_ih     = (const float*)d_in[9];
  const float* w_hh     = (const float*)d_in[10];
  const float* b_ih     = (const float*)d_in[11];
  const float* b_hh     = (const float*)d_in[12];
  const float* o1w      = (const float*)d_in[13];
  const float* o1b      = (const float*)d_in[14];
  const float* o2w      = (const float*)d_in[15];
  const float* o2b      = (const float*)d_in[16];
  float* out = (float*)d_out;

  const size_t MBT = (size_t)B_ * T_;              // 32768

  char* ws = (char*)d_ws;
  size_t off = 0;
  auto wsAlloc = [&](size_t bytes) -> void* {
    off = (off + 255) & ~(size_t)255;
    void* p = ws + off;
    off += bytes;
    return p;
  };

  bf16_t* xb    = (bf16_t*)wsAlloc(MBT * IN_ * 2);
  bf16_t* fcwb  = (bf16_t*)wsAlloc((size_t)H_ * IN_ * 2);
  bf16_t* ipwb  = (bf16_t*)wsAlloc((size_t)3 * H_ * H_ * 2);
  bf16_t* opwb  = (bf16_t*)wsAlloc((size_t)H_ * H_ * 2);
  bf16_t* wihb  = (bf16_t*)wsAlloc((size_t)L_ * 4 * H_ * H_ * 2);
  bf16_t* whhb  = (bf16_t*)wsAlloc((size_t)L_ * 4 * H_ * H_ * 2);
  bf16_t* o1wb  = (bf16_t*)wsAlloc((size_t)(H_ / 2) * H_ * 2);
  bf16_t* o2wb  = (bf16_t*)wsAlloc((size_t)OUT_ * (H_ / 2) * 2);
  bf16_t* hb    = (bf16_t*)wsAlloc(MBT * H_ * 2);
  bf16_t* qkvb  = (bf16_t*)wsAlloc(MBT * 3 * H_ * 2);
  bf16_t* ctxb  = (bf16_t*)wsAlloc(MBT * H_ * 2);
  bf16_t* seq0  = (bf16_t*)wsAlloc(MBT * H_ * 2);  // attn_out, [T,B,H] rows
  bf16_t* seq1  = (bf16_t*)wsAlloc(MBT * H_ * 2);  // layer-1 output
  bf16_t* seq2  = (bf16_t*)wsAlloc(MBT * H_ * 2);  // layer-2 output (lstm_out)
  float*  xg    = (float*)wsAlloc(MBT * 4 * H_ * 4);
  float*  hg    = (float*)wsAlloc((size_t)B_ * 4 * H_ * 4);
  bf16_t* hbf   = (bf16_t*)wsAlloc((size_t)B_ * H_ * 2);
  float*  cst   = (float*)wsAlloc((size_t)B_ * H_ * 4);
  bf16_t* y1b   = (bf16_t*)wsAlloc(MBT * (H_ / 2) * 2);
  float*  logit = (float*)wsAlloc(MBT * OUT_ * 4);

  auto cvt = [&](const float* s, bf16_t* d, size_t n) {
    cvt_k<<<2048, 256, 0, stream>>>(s, d, n);
  };
  auto gemm = [&](const bf16_t* A, const bf16_t* W, const float* bb0, const float* bb1,
                  float* Cf, bf16_t* Cb, int M, int N, int K, int relu, int perm) {
    dim3 g(N / TN, M / TM);
    gemm_bf16_k<<<g, 256, 0, stream>>>(A, W, bb0, bb1, Cf, Cb, M, N, K, relu, perm);
  };

  // ---- precision conversion (fp32 inputs -> bf16 WMMA operands) ----
  cvt(x,    xb,   MBT * IN_);
  cvt(fc_w, fcwb, (size_t)H_ * IN_);
  cvt(ipw,  ipwb, (size_t)3 * H_ * H_);
  cvt(opw,  opwb, (size_t)H_ * H_);
  cvt(w_ih, wihb, (size_t)L_ * 4 * H_ * H_);
  cvt(w_hh, whhb, (size_t)L_ * 4 * H_ * H_);
  cvt(o1w,  o1wb, (size_t)(H_ / 2) * H_);
  cvt(o2w,  o2wb, (size_t)OUT_ * (H_ / 2));

  // ---- fc: h = relu(x @ fc_w^T + b) ----
  gemm(xb, fcwb, fc_b, nullptr, nullptr, hb, (int)MBT, H_, IN_, 1, 0);
  // ---- qkv = h @ in_proj^T + b ----
  gemm(hb, ipwb, ipb, nullptr, nullptr, qkvb, (int)MBT, 3 * H_, H_, 0, 0);
  // ---- causal MHA ----
  hipFuncSetAttribute((const void*)attn_k, hipFuncAttributeMaxDynamicSharedMemorySize, ATTN_SMEM);
  attn_k<<<B_ * NH_, 256, ATTN_SMEM, stream>>>(qkvb, ctxb);
  // ---- attn_out = ctx @ out_proj^T + b, written as [T,B,H] for the LSTM ----
  gemm(ctxb, opwb, opb, nullptr, nullptr, seq0, (int)MBT, H_, H_, 0, 1);

  // ---- LSTM: xg = seq @ Wi^T + bi + bh hoisted; per-step recurrent WMMA GEMM ----
  const size_t outG = (size_t)B_ * T_ * OUT_;      // game_out offset in d_out
  const size_t outU = outG + (size_t)L_ * B_ * H_; // user_out offset
  for (int l = 0; l < L_; ++l) {
    const bf16_t* seqin = (l == 0) ? seq0 : seq1;
    bf16_t* seqout      = (l == 0) ? seq1 : seq2;
    gemm(seqin, wihb + (size_t)l * 4 * H_ * H_, b_ih + (size_t)l * 4 * H_,
         b_hh + (size_t)l * 4 * H_, xg, nullptr, (int)MBT, 4 * H_, H_, 0, 0);
    lstm_init_k<<<(B_ * H_) / 256, 256, 0, stream>>>(game_vec, user_vec, hbf, cst, l);
    for (int t = 0; t < T_; ++t) {
      gemm(hbf, whhb + (size_t)l * 4 * H_ * H_, nullptr, nullptr, hg, nullptr,
           B_, 4 * H_, H_, 0, 0);
      const bool last = (t == T_ - 1);
      lstm_point_k<<<(B_ * H_) / 256, 256, 0, stream>>>(
          xg, hg, cst, hbf, seqout,
          last ? out + outG + (size_t)l * B_ * H_ : nullptr,
          last ? out + outU + (size_t)l * B_ * H_ : nullptr, t);
    }
  }

  // ---- head: relu(linear) -> linear -> log_softmax ----
  gemm(seq2, o1wb, o1b, nullptr, nullptr, y1b, (int)MBT, H_ / 2, H_, 1, 0);
  gemm(y1b, o2wb, o2b, nullptr, logit, nullptr, (int)MBT, OUT_, H_ / 2, 0, 0);
  logsoftmax_k<<<(int)MBT, 128, 0, stream>>>(logit, out);
}
